// SchEmb_20109036879996
// MI455X (gfx1250) — compile-verified, matrix operand order checked by reading, not run on hardware
//
#include <hip/hip_runtime.h>
#include <math.h>

// ---------------- types ----------------
typedef __attribute__((ext_vector_type(16))) __bf16 v16bf;
typedef __attribute__((ext_vector_type(8)))  __bf16 v8bf;
typedef __attribute__((ext_vector_type(8)))  float  v8f;

#define NNODE 16384
#define NEDGE 262144
#define NG    128
#define NLAYER 5
#define LOG2F_C 0.6931471805599453f

// ---------------- device helpers ----------------
__device__ __forceinline__ v8f wmma_bf16(v16bf a, v16bf b, v8f c) {
    return __builtin_amdgcn_wmma_f32_16x16x32_bf16(false, a, false, b, (short)0, c, false, false);
}

// Load a 16-bit A/B operand fragment for v_wmma_*_16x16x32 from a row-major
// matrix (row stride 256 elements). Per ISA layout: lane L holds row (L&15);
// VGPR halves carry K chunks [kbase + half*8 .. +7] and [kbase+16+half*8 .. +7].
__device__ __forceinline__ v16bf load_frag_bf16(const __bf16* rowbase, int kbase, int half) {
    v8bf lo = *reinterpret_cast<const v8bf*>(rowbase + kbase + half * 8);
    v8bf hi = *reinterpret_cast<const v8bf*>(rowbase + kbase + 16 + half * 8);
    return __builtin_shufflevector(lo, hi, 0,1,2,3,4,5,6,7,8,9,10,11,12,13,14,15);
}

__device__ __forceinline__ v16bf load_frag_f32(const float* rowbase, int kbase, int half) {
    v8f lo = *reinterpret_cast<const v8f*>(rowbase + kbase + half * 8);
    v8f hi = *reinterpret_cast<const v8f*>(rowbase + kbase + 16 + half * 8);
    v16bf r;
#pragma unroll
    for (int i = 0; i < 8; ++i) { r[i] = (__bf16)lo[i]; r[8 + i] = (__bf16)hi[i]; }
    return r;
}

__device__ __forceinline__ float ssp(float v) {
    // SchNet shifted softplus: log(1+exp(v)) - log(2)
    float sp = (v > 20.f) ? v : log1pf(__expf(v));
    return sp - LOG2F_C;
}

__device__ __forceinline__ float gelu_exact(float v) {
    return 0.5f * v * (1.0f + erff(v * 0.70710678118654752f));
}

// ---------------- kernels ----------------

// x = concat(vert_emb[ids], pos@pos_emb, a_emb[ar]); also bf16 copy; zero agg.
__global__ void k_embed(const int* __restrict__ x_ids, const float* __restrict__ pos,
                        const int* __restrict__ ar,
                        const float* __restrict__ vert_emb, const float* __restrict__ pos_emb,
                        const float* __restrict__ a_emb,
                        float* __restrict__ x, __bf16* __restrict__ xbf, float* __restrict__ agg) {
    int t = blockIdx.x * blockDim.x + threadIdx.x;
    int n = t >> 8, c = t & 255;
    float v;
    if (c < 176) {
        v = vert_emb[x_ids[n] * 176 + c];
    } else if (c < 224) {
        int p = c - 176;
        v = pos[n * 3 + 0] * pos_emb[p] + pos[n * 3 + 1] * pos_emb[48 + p] +
            pos[n * 3 + 2] * pos_emb[96 + p];
    } else {
        v = a_emb[ar[n] * 32 + (c - 224)];
    }
    x[t] = v;
    xbf[t] = (__bf16)v;
    agg[t] = 0.f;
}

// 256x256 f32 [K][N] -> bf16 [N][K]
__global__ void k_transpose(const float* __restrict__ src, __bf16* __restrict__ dst) {
    int t = blockIdx.x * blockDim.x + threadIdx.x;
    int k = t >> 8, n = t & 255;
    dst[n * 256 + k] = (__bf16)src[k * 256 + n];
}

// h = x_bf @ lin1  (WMMA GEMM, one wave per 32 rows = 2 A row-tiles;
// each B fragment feeds 2 WMMAs -> half the weight-side L2 traffic)
__global__ void __launch_bounds__(256) k_lin1(const __bf16* __restrict__ xbf,
                                              const __bf16* __restrict__ wT,
                                              float* __restrict__ h) {
    int lane = threadIdx.x & 31, wave = threadIdx.x >> 5;
    int row0 = (blockIdx.x * 8 + wave) * 32;
    int m = lane & 15, half = lane >> 4;

    const __bf16* arow0 = xbf + (size_t)(row0 + m) * 256;
    const __bf16* arow1 = xbf + (size_t)(row0 + 16 + m) * 256;
    v16bf a0[8], a1[8];
#pragma unroll
    for (int k = 0; k < 8; ++k) {
        a0[k] = load_frag_bf16(arow0, k * 32, half);
        a1[k] = load_frag_bf16(arow1, k * 32, half);
    }

    for (int nt = 0; nt < 16; ++nt) {
        int col = nt * 16 + m;
        const __bf16* brow = wT + (size_t)col * 256;
        // pull next nt's weight rows toward the WGP while this tile computes
        __builtin_prefetch(brow + 16 * 256, 0, 3);
        v8f acc0 = {}, acc1 = {};
#pragma unroll
        for (int k = 0; k < 8; ++k) {
            v16bf b = load_frag_bf16(brow, k * 32, half);
            acc0 = wmma_bf16(a0[k], b, acc0);
            acc1 = wmma_bf16(a1[k], b, acc1);
        }
#pragma unroll
        for (int v = 0; v < 8; ++v) {
            h[(size_t)(row0 + v + half * 8) * 256 + col] = acc0[v];
            h[(size_t)(row0 + 16 + v + half * 8) * 256 + col] = acc1[v];
        }
    }
}

// Edge pipeline, 32 edges per wave (2 row-tiles), 2 waves/block:
// ea = attr@edge_emb (regs) ; T1 = ssp(ea@W1+b1) (LDS round-trip) ;
// W = T1@W2+b2 ; msg = h[src]*W ; atomicAdd(agg[dst], msg).
__global__ void __launch_bounds__(64) k_edge(const float* __restrict__ edge_attr,
                                             const float* __restrict__ edge_emb_g,
                                             const int* __restrict__ esrc, const int* __restrict__ edst,
                                             const float* __restrict__ h,
                                             const __bf16* __restrict__ w1T, const float* __restrict__ b1,
                                             const __bf16* __restrict__ w2T, const float* __restrict__ b2,
                                             float* __restrict__ agg) {
    __shared__ float eemb[4 * 256];
    __shared__ __bf16 T1[2][32 * 256];
    for (int i = threadIdx.x; i < 1024; i += 64) eemb[i] = edge_emb_g[i];
    __syncthreads();

    int lane = threadIdx.x & 31, wave = threadIdx.x >> 5;
    int e0 = (blockIdx.x * 2 + wave) * 32;
    int m = lane & 15, half = lane >> 4;

    const float4 at0 = reinterpret_cast<const float4*>(edge_attr)[e0 + m];
    const float4 at1 = reinterpret_cast<const float4*>(edge_attr)[e0 + 16 + m];

    // A fragments of ea for both row-tiles (computed on the fly -> bf16)
    v16bf a0[8], a1[8];
#pragma unroll
    for (int k = 0; k < 8; ++k) {
        v16bf t0, t1v;
#pragma unroll
        for (int i = 0; i < 8; ++i) {
            int c0 = k * 32 + i + half * 8;
            int c1 = c0 + 16;
            float e00 = eemb[c0], e10 = eemb[256 + c0], e20 = eemb[512 + c0], e30 = eemb[768 + c0];
            float e01 = eemb[c1], e11 = eemb[256 + c1], e21 = eemb[512 + c1], e31 = eemb[768 + c1];
            t0[i]     = (__bf16)(at0.x * e00 + at0.y * e10 + at0.z * e20 + at0.w * e30);
            t0[8 + i] = (__bf16)(at0.x * e01 + at0.y * e11 + at0.z * e21 + at0.w * e31);
            t1v[i]     = (__bf16)(at1.x * e00 + at1.y * e10 + at1.z * e20 + at1.w * e30);
            t1v[8 + i] = (__bf16)(at1.x * e01 + at1.y * e11 + at1.z * e21 + at1.w * e31);
        }
        a0[k] = t0;
        a1[k] = t1v;
    }

    __bf16* t1 = &T1[wave][0];

    // Stage 1: T1 = bf16(ssp(ea @ W1 + b1))
    for (int nt = 0; nt < 16; ++nt) {
        int col = nt * 16 + m;
        const __bf16* brow = w1T + (size_t)col * 256;
        __builtin_prefetch(brow + 16 * 256, 0, 3);
        v8f acc0 = {}, acc1 = {};
#pragma unroll
        for (int k = 0; k < 8; ++k) {
            v16bf b = load_frag_bf16(brow, k * 32, half);
            acc0 = wmma_bf16(a0[k], b, acc0);
            acc1 = wmma_bf16(a1[k], b, acc1);
        }
        float bias = b1[col];
#pragma unroll
        for (int v = 0; v < 8; ++v) {
            t1[(v + half * 8) * 256 + col] = (__bf16)ssp(acc0[v] + bias);
            t1[(16 + v + half * 8) * 256 + col] = (__bf16)ssp(acc1[v] + bias);
        }
    }
    __syncthreads();

    // src/dst for the 16 edge rows this lane owns in C-layout
    int srcs0[8], dsts0[8], srcs1[8], dsts1[8];
#pragma unroll
    for (int v = 0; v < 8; ++v) {
        int ea_ = e0 + v + half * 8;
        int eb_ = e0 + 16 + v + half * 8;
        srcs0[v] = esrc[ea_]; dsts0[v] = edst[ea_];
        srcs1[v] = esrc[eb_]; dsts1[v] = edst[eb_];
    }

    // Stage 2: W = T1 @ W2 + b2, fused gather-multiply-scatter.
    // Hoist T1 A-fragments: stage-1 a0/a1 are dead, registers are reused.
    const __bf16* a2row0 = t1 + m * 256;
    const __bf16* a2row1 = t1 + (16 + m) * 256;
#pragma unroll
    for (int k = 0; k < 8; ++k) {
        a0[k] = load_frag_bf16(a2row0, k * 32, half);
        a1[k] = load_frag_bf16(a2row1, k * 32, half);
    }
    for (int nt = 0; nt < 16; ++nt) {
        int col = nt * 16 + m;
        const __bf16* brow = w2T + (size_t)col * 256;
        __builtin_prefetch(brow + 16 * 256, 0, 3);
        v8f acc0 = {}, acc1 = {};
#pragma unroll
        for (int k = 0; k < 8; ++k) {
            v16bf b = load_frag_bf16(brow, k * 32, half);
            acc0 = wmma_bf16(a0[k], b, acc0);
            acc1 = wmma_bf16(a1[k], b, acc1);
        }
        float bias = b2[col];
#pragma unroll
        for (int v = 0; v < 8; ++v) {
            float W0 = acc0[v] + bias;
            float W1 = acc1[v] + bias;
            atomicAdd(&agg[(size_t)dsts0[v] * 256 + col], h[(size_t)srcs0[v] * 256 + col] * W0);
            atomicAdd(&agg[(size_t)dsts1[v] * 256 + col], h[(size_t)srcs1[v] * 256 + col] * W1);
        }
    }
}

// conv = ssp(agg@lin2+b2)@lin+b ; x = relu(conv)+x ; refresh xbf ; zero agg.
// 32 nodes per wave, 2 waves/block.
__global__ void __launch_bounds__(64) k_conv(float* __restrict__ agg,
                                             const __bf16* __restrict__ lin2T, const float* __restrict__ b2,
                                             const __bf16* __restrict__ linT, const float* __restrict__ bl,
                                             float* x, __bf16* __restrict__ xbf) {
    __shared__ __bf16 T1[2][32 * 256];
    int lane = threadIdx.x & 31, wave = threadIdx.x >> 5;
    int row0 = (blockIdx.x * 2 + wave) * 32;
    int m = lane & 15, half = lane >> 4;

    const float* arow0 = agg + (size_t)(row0 + m) * 256;
    const float* arow1 = agg + (size_t)(row0 + 16 + m) * 256;
    v16bf a0[8], a1[8];
#pragma unroll
    for (int k = 0; k < 8; ++k) {
        a0[k] = load_frag_f32(arow0, k * 32, half);
        a1[k] = load_frag_f32(arow1, k * 32, half);
    }

    __bf16* t1 = &T1[wave][0];

    for (int nt = 0; nt < 16; ++nt) {
        int col = nt * 16 + m;
        const __bf16* brow = lin2T + (size_t)col * 256;
        __builtin_prefetch(brow + 16 * 256, 0, 3);
        v8f acc0 = {}, acc1 = {};
#pragma unroll
        for (int k = 0; k < 8; ++k) {
            v16bf b = load_frag_bf16(brow, k * 32, half);
            acc0 = wmma_bf16(a0[k], b, acc0);
            acc1 = wmma_bf16(a1[k], b, acc1);
        }
        float bias = b2[col];
#pragma unroll
        for (int v = 0; v < 8; ++v) {
            t1[(v + half * 8) * 256 + col] = (__bf16)ssp(acc0[v] + bias);
            t1[(16 + v + half * 8) * 256 + col] = (__bf16)ssp(acc1[v] + bias);
        }
    }
    __syncthreads();

    // Hoisted stage-2 A fragments (reuse a0/a1 registers)
    const __bf16* a2row0 = t1 + m * 256;
    const __bf16* a2row1 = t1 + (16 + m) * 256;
#pragma unroll
    for (int k = 0; k < 8; ++k) {
        a0[k] = load_frag_bf16(a2row0, k * 32, half);
        a1[k] = load_frag_bf16(a2row1, k * 32, half);
    }
    for (int nt = 0; nt < 16; ++nt) {
        int col = nt * 16 + m;
        const __bf16* brow = linT + (size_t)col * 256;
        __builtin_prefetch(brow + 16 * 256, 0, 3);
        v8f acc0 = {}, acc1 = {};
#pragma unroll
        for (int k = 0; k < 8; ++k) {
            v16bf b = load_frag_bf16(brow, k * 32, half);
            acc0 = wmma_bf16(a0[k], b, acc0);
            acc1 = wmma_bf16(a1[k], b, acc1);
        }
        float bias = bl[col];
#pragma unroll
        for (int v = 0; v < 8; ++v) {
            size_t i0 = (size_t)(row0 + v + half * 8) * 256 + col;
            size_t i1 = (size_t)(row0 + 16 + v + half * 8) * 256 + col;
            float nx0 = x[i0] + fmaxf(acc0[v] + bias, 0.f);
            float nx1 = x[i1] + fmaxf(acc1[v] + bias, 0.f);
            x[i0] = nx0; xbf[i0] = (__bf16)nx0; agg[i0] = 0.f;
            x[i1] = nx1; xbf[i1] = (__bf16)nx1; agg[i1] = 0.f;
        }
    }
}

__global__ void k_pool_zero(float* __restrict__ pooled, float* __restrict__ cnt) {
    int t = blockIdx.x * blockDim.x + threadIdx.x;
    if (t < NG * 256) pooled[t] = 0.f;
    else if (t < NG * 256 + NG) cnt[t - NG * 256] = 0.f;
}

__global__ void k_pool(const float* __restrict__ x, const int* __restrict__ batch,
                       float* __restrict__ pooled, float* __restrict__ cnt) {
    int t = blockIdx.x * blockDim.x + threadIdx.x;
    int n = t >> 8, c = t & 255;
    int b = batch[n];
    atomicAdd(&pooled[b * 256 + c], x[t]);
    if (c == 0) atomicAdd(&cnt[b], 1.f);
}

__global__ void k_head1(const float* __restrict__ pooled, const float* __restrict__ cnt,
                        const float* __restrict__ w1, const float* __restrict__ b1,
                        float* __restrict__ hd) {
    int t = blockIdx.x * blockDim.x + threadIdx.x;
    int g = t >> 9, j = t & 511;
    float inv = 1.f / fmaxf(cnt[g], 1.f);
    float s = 0.f;
    for (int k = 0; k < 256; ++k)
        s += pooled[g * 256 + k] * inv * w1[k * 512 + j];
    hd[t] = gelu_exact(s + b1[j]);
}

__global__ void k_head2(const float* __restrict__ hd, const float* __restrict__ w2,
                        const float* __restrict__ b2, float* __restrict__ out) {
    __shared__ float red[256];
    int g = blockIdx.x, t = threadIdx.x;
    red[t] = hd[g * 512 + t] * w2[t] + hd[g * 512 + 256 + t] * w2[256 + t];
    __syncthreads();
    for (int o = 128; o > 0; o >>= 1) {
        if (t < o) red[t] += red[t + o];
        __syncthreads();
    }
    if (t == 0) out[g] = red[0] + b2[0];
}

// ---------------- host ----------------
extern "C" void kernel_launch(void* const* d_in, const int* in_sizes, int n_in,
                              void* d_out, int out_size, void* d_ws, size_t ws_size,
                              hipStream_t stream) {
    (void)in_sizes; (void)n_in; (void)out_size; (void)ws_size;

    const int*   x_ids    = (const int*)d_in[0];
    const int*   eindex   = (const int*)d_in[1];
    const float* eattr    = (const float*)d_in[2];
    const int*   batch    = (const int*)d_in[3];
    const float* pos      = (const float*)d_in[4];
    const int*   ar       = (const int*)d_in[5];
    const float* vert_emb = (const float*)d_in[6];
    const float* pos_emb  = (const float*)d_in[7];
    const float* a_emb    = (const float*)d_in[8];
    const float* edge_emb = (const float*)d_in[9];
    const float* mlp_w1   = (const float*)d_in[10];
    const float* mlp_b1   = (const float*)d_in[11];
    const float* mlp_w2   = (const float*)d_in[12];
    const float* mlp_b2   = (const float*)d_in[13];
    const float* lin1_w   = (const float*)d_in[14];
    const float* lin2_w   = (const float*)d_in[15];
    const float* lin2_b   = (const float*)d_in[16];
    const float* lin_w    = (const float*)d_in[17];
    const float* lin_b    = (const float*)d_in[18];
    const float* head_w1  = (const float*)d_in[19];
    const float* head_b1  = (const float*)d_in[20];
    const float* head_w2  = (const float*)d_in[21];
    const float* head_b2  = (const float*)d_in[22];

    const int* esrc = eindex;
    const int* edst = eindex + NEDGE;

    // workspace carve-up
    char* ws = (char*)d_ws;
    size_t off = 0;
    float*  x      = (float*)(ws + off);  off += (size_t)NNODE * 256 * 4;
    __bf16* xbf    = (__bf16*)(ws + off); off += (size_t)NNODE * 256 * 2;
    float*  h      = (float*)(ws + off);  off += (size_t)NNODE * 256 * 4;
    float*  agg    = (float*)(ws + off);  off += (size_t)NNODE * 256 * 4;
    __bf16* wT     = (__bf16*)(ws + off); off += (size_t)25 * 65536 * 2;
    float*  pooled = (float*)(ws + off);  off += (size_t)NG * 256 * 4;
    float*  cnt    = (float*)(ws + off);  off += 512;
    float*  hd     = (float*)(ws + off);  off += (size_t)NG * 512 * 4;

    // 1) embeddings + zero agg
    k_embed<<<NNODE * 256 / 256, 256, 0, stream>>>(x_ids, pos, ar, vert_emb, pos_emb, a_emb,
                                                   x, xbf, agg);

    // 2) transpose all per-layer 256x256 weights to bf16 [N][K]
    const float* srcs[5] = {mlp_w1, mlp_w2, lin1_w, lin2_w, lin_w};
    for (int l = 0; l < NLAYER; ++l)
        for (int mm = 0; mm < 5; ++mm)
            k_transpose<<<256, 256, 0, stream>>>(srcs[mm] + (size_t)l * 65536,
                                                 wT + (size_t)(l * 5 + mm) * 65536);

    // 3) layers
    for (int l = 0; l < NLAYER; ++l) {
        const __bf16* w1T   = wT + (size_t)(l * 5 + 0) * 65536;
        const __bf16* w2T   = wT + (size_t)(l * 5 + 1) * 65536;
        const __bf16* lin1T = wT + (size_t)(l * 5 + 2) * 65536;
        const __bf16* lin2T = wT + (size_t)(l * 5 + 3) * 65536;
        const __bf16* linT  = wT + (size_t)(l * 5 + 4) * 65536;

        k_lin1<<<NNODE / 32 / 8, 256, 0, stream>>>(xbf, lin1T, h);
        k_edge<<<NEDGE / 32 / 2, 64, 0, stream>>>(eattr, edge_emb, esrc, edst, h,
                                                  w1T, mlp_b1 + l * 256,
                                                  w2T, mlp_b2 + l * 256, agg);
        k_conv<<<NNODE / 32 / 2, 64, 0, stream>>>(agg, lin2T, lin2_b + l * 256,
                                                  linT, lin_b + l * 256, x, xbf);
    }

    // 4) pool + head
    k_pool_zero<<<(NG * 256 + NG + 255) / 256, 256, 0, stream>>>(pooled, cnt);
    k_pool<<<NNODE * 256 / 256, 256, 0, stream>>>(x, batch, pooled, cnt);
    k_head1<<<NG * 512 / 256, 256, 0, stream>>>(pooled, cnt, head_w1, head_b1, hd);
    k_head2<<<NG, 256, 0, stream>>>(hd, head_w2, head_b2, (float*)d_out);
}
